// GSL_mlp_9698036155128
// MI455X (gfx1250) — matrix operand order repaired and checked by reading, not compiled, not run on hardware
//
#include <hip/hip_runtime.h>
#include <stdint.h>

typedef __attribute__((ext_vector_type(16))) _Float16 v16h;
typedef __attribute__((ext_vector_type(8)))  float    v8f;

namespace {

constexpr int NN = 100;            // nodes
constexpr int BB = 8;              // batch
constexpr int HH = 256;            // hidden
constexpr int EE = NN * (NN - 1);  // 9900 edges
constexpr int ME = BB * EE;        // 79200 edge rows
constexpr int MN = BB * NN;        // 800 node rows

union AV { uint32_t u[8]; _Float16 h[16]; v16h v; };

// One wave computes one 16x16 C tile: C = act(A[M,K] * Wt[N,K]^T + bias)
// AMODE 0: A from f16 row-major buffer A16
// AMODE 1: A is virtual node2edge(Gn) -> [ME, 2H], Gn = bn-applied f16 [MN,H]
// AMODE 2: A is virtual [node2edge(Gn) | bn(Hk)] -> [ME, 3H]
template <int AMODE>
__global__ __launch_bounds__(128)
void wmma_gemm(const _Float16* __restrict__ A16,
               const _Float16* __restrict__ Gn,
               const float* __restrict__ Hk, const float* __restrict__ sck,
               const float* __restrict__ shk,
               const _Float16* __restrict__ Wt,  // [N,K] f16 (pre-transposed)
               const float* __restrict__ bias,
               _Float16* __restrict__ O16, float* __restrict__ O32,
               int M, int N, int K, int relu) {
  const int lane   = threadIdx.x & 31;
  const int wave   = threadIdx.x >> 5;
  const int tilesN = N >> 4;
  const int ntiles = (M >> 4) * tilesN;
  const int t = blockIdx.x * 4 + wave;
  if (t >= ntiles) return;            // wave-uniform: EXEC stays all-ones
  const int tm = t / tilesN, tn = t % tilesN;

  const int lhalf = lane >> 4;        // 0 / 1
  const int lmod  = lane & 15;
  const int koffA = lhalf << 3;       // A: lanes 16-31 hold K+8 block
  const int koffB = lhalf << 3;       // B: lanes 16-31 hold K+8 block
  const int am    = tm * 16 + lmod;   // A row for this lane
  const int ncol  = tn * 16 + lmod;   // B column / C column

  const _Float16* gs = nullptr;       // sender node feature row (f16, bn applied)
  const _Float16* gr = nullptr;       // receiver node feature row
  if (AMODE != 0) {
    // decode edge row: e = i*(N-1) + (j<i ? j : j-1); REC=i, SEND=j
    int b  = am / EE;
    int e  = am - b * EE;
    int i  = e / (NN - 1);
    int jj = e - i * (NN - 1);
    int j  = jj + (jj >= i ? 1 : 0);
    gs = Gn + (size_t)(b * NN + j) * HH;
    gr = Gn + (size_t)(b * NN + i) * HH;
  }

  v8f acc = {};
  const uint32_t* bp_base =
      (const uint32_t*)(Wt + (size_t)ncol * K + koffB);
  const uint32_t* ap_base = (AMODE == 0)
      ? (const uint32_t*)(A16 + (size_t)am * K + koffA) : nullptr;

  for (int kc = 0; kc < K; kc += 32) {
    AV a, b;
    // B operand: halfs 0-7 = K[koffB..koffB+7], halfs 8-15 = K[+16..+23]
    const uint32_t* bp = bp_base + (kc >> 1);
#pragma unroll
    for (int p = 0; p < 4; ++p) { b.u[p] = bp[p]; b.u[4 + p] = bp[8 + p]; }

    if (AMODE == 0) {
      // A operand: halfs 0-7 = K[koffA..koffA+7], halfs 8-15 = K[+16..+23]
      const uint32_t* ap = ap_base + (kc >> 1);
#pragma unroll
      for (int p = 0; p < 4; ++p) { a.u[p] = ap[p]; a.u[4 + p] = ap[8 + p]; }
    } else {
      // A 32-wide K-chunk never straddles a 256 boundary -> uniform region
      if (AMODE == 2 && kc >= 2 * HH) {
        // skip region: f32 H4-input row with fused bn2, convert to f16
        const int cb = kc - 2 * HH;
        const float* hb = Hk + (size_t)am * HH + cb;
#pragma unroll
        for (int p = 0; p < 8; ++p) {
          int c = koffA + ((p < 4) ? 2 * p : 16 + 2 * (p - 4));
          float v0 = hb[c]     * sck[cb + c]     + shk[cb + c];
          float v1 = hb[c + 1] * sck[cb + c + 1] + shk[cb + c + 1];
          a.h[2 * p]     = (_Float16)v0;
          a.h[2 * p + 1] = (_Float16)v1;
        }
      } else {
        const _Float16* base = (kc < HH) ? (gs + kc) : (gr + (kc - HH));
        const uint32_t* ap = (const uint32_t*)(base + koffA);
#pragma unroll
        for (int p = 0; p < 4; ++p) { a.u[p] = ap[p]; a.u[4 + p] = ap[8 + p]; }
      }
    }
    acc = __builtin_amdgcn_wmma_f32_16x16x32_f16(false, a.v, false, b.v,
                                                 (short)0, acc, false, false);
  }

  const float bv = bias[ncol];
#pragma unroll
  for (int g = 0; g < 8; ++g) {
    int m = tm * 16 + g + (lhalf << 3);   // C layout: lanes16-31 -> M+8
    float v = acc[g] + bv;
    if (relu) v = fmaxf(v, 0.0f);
    size_t idx = (size_t)m * N + ncol;
    if (O16) O16[idx] = (_Float16)v;
    if (O32) O32[idx] = v;
  }
}

// fp32 [K,N] -> fp16 [N,K] (weights, done once per launch)
__global__ void transpose_to_f16(const float* __restrict__ W,
                                 _Float16* __restrict__ Wt, int K, int N) {
  int gid = blockIdx.x * blockDim.x + threadIdx.x;
  if (gid >= K * N) return;
  int k = gid / N, n = gid - k * N;
  Wt[(size_t)n * K + k] = (_Float16)W[gid];
}

__global__ void f32_to_f16_vec(const float* __restrict__ in,
                               _Float16* __restrict__ out, int n) {
  int gid = blockIdx.x * blockDim.x + threadIdx.x;
  if (gid < n) out[gid] = (_Float16)in[gid];
}

// out_f16[r,c] = H[r,c]*sc[c] + sh[c]   (node buffers, M*256 elements)
__global__ void bn_apply_f16(const float* __restrict__ H,
                             const float* __restrict__ sc,
                             const float* __restrict__ sh,
                             _Float16* __restrict__ out, int n) {
  int gid = blockIdx.x * blockDim.x + threadIdx.x;
  if (gid >= n) return;
  int c = gid & (HH - 1);
  out[gid] = (_Float16)(H[gid] * sc[c] + sh[c]);
}

// Per-channel batch stats over M rows of H[M,256]; emit fused scale/shift.
__global__ void bn_stats(const float* __restrict__ H, int M,
                         const float* __restrict__ g,
                         const float* __restrict__ beta,
                         float* __restrict__ sc, float* __restrict__ sh) {
  int c = blockIdx.x, tid = threadIdx.x;
  double s = 0.0, s2 = 0.0;
  for (int r = tid; r < M; r += blockDim.x) {
    double v = (double)H[(size_t)r * HH + c];
    s += v; s2 += v * v;
  }
  __shared__ double ls[256], lq[256];
  ls[tid] = s; lq[tid] = s2;
  __syncthreads();
  for (int off = 128; off > 0; off >>= 1) {
    if (tid < off) { ls[tid] += ls[tid + off]; lq[tid] += lq[tid + off]; }
    __syncthreads();
  }
  if (tid == 0) {
    double mu  = ls[0] / M;
    double var = lq[0] / M - mu * mu;
    if (var < 0.0) var = 0.0;
    float scale = g[c] * rsqrtf((float)var + 1e-5f);
    sc[c] = scale;
    sh[c] = beta[c] - (float)mu * scale;
  }
}

// XN[b,i,c] = (1/100) * sum_{e: REC=i} bn2(H2[b,e,c]); edges with REC=i are
// contiguous [99i, 99i+99). Affine BN commutes: raw-sum*sc/100 + 0.99*sh.
__global__ void edge2node_bn(const float* __restrict__ H2,
                             const float* __restrict__ sc,
                             const float* __restrict__ sh,
                             _Float16* __restrict__ XN) {
  int c  = threadIdx.x;       // 256
  int bi = blockIdx.x;        // 0..799
  int b = bi / NN, i = bi - b * NN;
  size_t base = ((size_t)b * EE + (size_t)i * (NN - 1)) * HH + c;
  float acc = 0.f;
  for (int k = 0; k < NN - 1; ++k) acc += H2[base + (size_t)k * HH];
  float v = acc * 0.01f * sc[c] + 0.99f * sh[c];
  XN[(size_t)bi * HH + c] = (_Float16)v;
}

// out[m,o] = fc_b[o] + sum_c bn4(H4[m,c]) * fc_w[c,o]  (OUT=4, VALU path)
__global__ void fc_head(const float* __restrict__ H4,
                        const float* __restrict__ sc,
                        const float* __restrict__ sh,
                        const float* __restrict__ W,
                        const float* __restrict__ bfc,
                        float* __restrict__ out, int M) {
  int gid = blockIdx.x * blockDim.x + threadIdx.x;
  if (gid >= M * 4) return;
  int m = gid >> 2, o = gid & 3;
  const float* row = H4 + (size_t)m * HH;
  float acc = bfc[o];
  for (int c = 0; c < HH; ++c) acc += (row[c] * sc[c] + sh[c]) * W[c * 4 + o];
  out[gid] = acc;
}

inline int gemm_grid(int M, int N) { return ((M >> 4) * (N >> 4) + 3) / 4; }

}  // namespace

extern "C" void kernel_launch(void* const* d_in, const int* in_sizes, int n_in,
                              void* d_out, int out_size, void* d_ws, size_t ws_size,
                              hipStream_t stream) {
  const float* x    = (const float*)d_in[0];
  const float* m1w1 = (const float*)d_in[6];  const float* m1b1 = (const float*)d_in[7];
  const float* m1w2 = (const float*)d_in[8];  const float* m1b2 = (const float*)d_in[9];
  const float* m1g  = (const float*)d_in[10]; const float* m1be = (const float*)d_in[11];
  const float* m2w1 = (const float*)d_in[12]; const float* m2b1 = (const float*)d_in[13];
  const float* m2w2 = (const float*)d_in[14]; const float* m2b2 = (const float*)d_in[15];
  const float* m2g  = (const float*)d_in[16]; const float* m2be = (const float*)d_in[17];
  const float* m3w1 = (const float*)d_in[18]; const float* m3b1 = (const float*)d_in[19];
  const float* m3w2 = (const float*)d_in[20]; const float* m3b2 = (const float*)d_in[21];
  const float* m3g  = (const float*)d_in[22]; const float* m3be = (const float*)d_in[23];
  const float* m4w1 = (const float*)d_in[24]; const float* m4b1 = (const float*)d_in[25];
  const float* m4w2 = (const float*)d_in[26]; const float* m4b2 = (const float*)d_in[27];
  const float* m4g  = (const float*)d_in[28]; const float* m4be = (const float*)d_in[29];
  const float* fcw  = (const float*)d_in[30]; const float* fcb  = (const float*)d_in[31];
  float* out = (float*)d_out;

  char* ws = (char*)d_ws;
  auto H16 = [&](size_t o) { return (_Float16*)(ws + o); };
  auto F32 = [&](size_t o) { return (float*)(ws + o); };

  // workspace map (bytes)
  const size_t oWt1a = 0;         // 256*256*2
  const size_t oWt1b = 131072;
  const size_t oWt2a = 262144;    // 512*256*2
  const size_t oWt2b = 524288;
  const size_t oWt3a = 655360;
  const size_t oWt3b = 786432;
  const size_t oWt4a = 917504;    // 768*256*2
  const size_t oWt4b = 1310720;
  const size_t oSc1 = 1441792, oSh1 = 1442816;
  const size_t oSc2 = 1443840, oSh2 = 1444864;
  const size_t oSc3 = 1445888, oSh3 = 1446912;
  const size_t oSc4 = 1447936, oSh4 = 1448960;
  const size_t oX0  = 1449984;    // 800*256*2
  const size_t oT1  = 1859584;    // 800*256*2
  const size_t oH1  = 2269184;    // 800*256*4
  const size_t oXN  = 3088384;    // 800*256*2
  const size_t oT3  = 3497984;    // 800*256*2
  const size_t oH3  = 3907584;    // 800*256*4
  const size_t oXb1 = 4726784;    // 800*256*2  bn1-applied node feats f16
  const size_t oXb3 = 5136384;    // 800*256*2  bn3-applied node feats f16
  const size_t oT2  = 5545984;    // 79200*256*2  (T4 aliases later)
  const size_t oH2  = 46096384;   // 79200*256*4  (H4 aliases later)

  // ---- weight prep: fp32 [K,N] -> fp16 [N,K]
  transpose_to_f16<<<256, 256, 0, stream>>>(m1w1, H16(oWt1a), 256, 256);
  transpose_to_f16<<<256, 256, 0, stream>>>(m1w2, H16(oWt1b), 256, 256);
  transpose_to_f16<<<512, 256, 0, stream>>>(m2w1, H16(oWt2a), 512, 256);
  transpose_to_f16<<<256, 256, 0, stream>>>(m2w2, H16(oWt2b), 256, 256);
  transpose_to_f16<<<256, 256, 0, stream>>>(m3w1, H16(oWt3a), 256, 256);
  transpose_to_f16<<<256, 256, 0, stream>>>(m3w2, H16(oWt3b), 256, 256);
  transpose_to_f16<<<768, 256, 0, stream>>>(m4w1, H16(oWt4a), 768, 256);
  transpose_to_f16<<<256, 256, 0, stream>>>(m4w2, H16(oWt4b), 256, 256);
  f32_to_f16_vec<<<(MN * HH + 255) / 256, 256, 0, stream>>>(x, H16(oX0), MN * HH);

  // ---- mlp1 (node rows, M=800)
  wmma_gemm<0><<<gemm_grid(MN, HH), 128, 0, stream>>>(
      H16(oX0), nullptr, nullptr, nullptr, nullptr,
      H16(oWt1a), m1b1, H16(oT1), nullptr, MN, HH, 256, 1);
  wmma_gemm<0><<<gemm_grid(MN, HH), 128, 0, stream>>>(
      H16(oT1), nullptr, nullptr, nullptr, nullptr,
      H16(oWt1b), m1b2, nullptr, F32(oH1), MN, HH, 256, 1);
  bn_stats<<<256, 256, 0, stream>>>(F32(oH1), MN, m1g, m1be, F32(oSc1), F32(oSh1));
  bn_apply_f16<<<MN, 256, 0, stream>>>(F32(oH1), F32(oSc1), F32(oSh1),
                                       H16(oXb1), MN * HH);

  // ---- mlp2 (edge rows, M=79200; A = node2edge(Xb1) fused gather, K=512)
  wmma_gemm<1><<<gemm_grid(ME, HH), 128, 0, stream>>>(
      nullptr, H16(oXb1), nullptr, nullptr, nullptr,
      H16(oWt2a), m2b1, H16(oT2), nullptr, ME, HH, 512, 1);
  wmma_gemm<0><<<gemm_grid(ME, HH), 128, 0, stream>>>(
      H16(oT2), nullptr, nullptr, nullptr, nullptr,
      H16(oWt2b), m2b2, nullptr, F32(oH2), ME, HH, 256, 1);
  bn_stats<<<256, 256, 0, stream>>>(F32(oH2), ME, m2g, m2be, F32(oSc2), F32(oSh2));

  // ---- edge2node (bn2 fused post-sum; contiguous REC runs) -> XN f16
  edge2node_bn<<<MN, 256, 0, stream>>>(F32(oH2), F32(oSc2), F32(oSh2), H16(oXN));

  // ---- mlp3 (node rows)
  wmma_gemm<0><<<gemm_grid(MN, HH), 128, 0, stream>>>(
      H16(oXN), nullptr, nullptr, nullptr, nullptr,
      H16(oWt3a), m3b1, H16(oT3), nullptr, MN, HH, 256, 1);
  wmma_gemm<0><<<gemm_grid(MN, HH), 128, 0, stream>>>(
      H16(oT3), nullptr, nullptr, nullptr, nullptr,
      H16(oWt3b), m3b2, nullptr, F32(oH3), MN, HH, 256, 1);
  bn_stats<<<256, 256, 0, stream>>>(F32(oH3), MN, m3g, m3be, F32(oSc3), F32(oSh3));
  bn_apply_f16<<<MN, 256, 0, stream>>>(F32(oH3), F32(oSc3), F32(oSh3),
                                       H16(oXb3), MN * HH);

  // ---- mlp4 (A = [node2edge(Xb3) | bn2(H2)] fused, K=768); T4 aliases T2
  wmma_gemm<2><<<gemm_grid(ME, HH), 128, 0, stream>>>(
      nullptr, H16(oXb3), F32(oH2), F32(oSc2), F32(oSh2),
      H16(oWt4a), m4b1, H16(oT2), nullptr, ME, HH, 768, 1);
  // H4 aliases H2 (H2 dead after previous kernel)
  wmma_gemm<0><<<gemm_grid(ME, HH), 128, 0, stream>>>(
      H16(oT2), nullptr, nullptr, nullptr, nullptr,
      H16(oWt4b), m4b2, nullptr, F32(oH2), ME, HH, 256, 1);
  bn_stats<<<256, 256, 0, stream>>>(F32(oH2), ME, m4g, m4be, F32(oSc4), F32(oSh4));

  // ---- fc head: bn4 fused, OUT=4
  fc_head<<<(ME * 4 + 255) / 256, 256, 0, stream>>>(
      F32(oH2), F32(oSc4), F32(oSh4), fcw, fcb, out, ME);
}
